// LocalSumMessageFunction_17025250362097
// MI455X (gfx1250) — matrix operand order repaired and checked by reading, not compiled, not run on hardware
//
#include <hip/hip_runtime.h>
#include <hip/hip_bf16.h>
#include <math.h>

// Problem constants (match reference)
constexpr int N_NODES = 100000;
constexpr int N_EDGES = 800000;
constexpr int D_COORD = 16;
constexpr int N_FEAT  = 8;
constexpr int D_IN    = 40;   // 8 + 16 + 16 (padded to 64 in LDS)
constexpr int D_OUT   = 16;

typedef _Float16 v16h __attribute__((ext_vector_type(16)));
typedef _Float16 v8h  __attribute__((ext_vector_type(8)));
typedef float    v8f  __attribute__((ext_vector_type(8)));

#define WPB 8            // waves per block
#define BLOCK (WPB * 32)

// Pack two f32 into one dword of two f16 (round-to-nearest via scalar cvt)
static __device__ inline unsigned pk2(float a, float b) {
    union { _Float16 h[2]; unsigned u; } x;
    x.h[0] = (_Float16)a; x.h[1] = (_Float16)b;
    return x.u;
}

// Build a 16x32 f16 A-fragment from an LDS row (row-major, 64 f16 per row).
// Lanes 0-15 pass kbase=koff    -> K = koff..+15
// Lanes 16-31 pass kbase=koff+16 -> K = koff+16..+31
static __device__ inline v16h loadA(const _Float16* rowp, int kbase) {
    v8h lo = *(const v8h*)(rowp + kbase);
    v8h hi = *(const v8h*)(rowp + kbase + 8);
    return __builtin_shufflevector(lo, hi, 0,1,2,3,4,5,6,7,8,9,10,11,12,13,14,15);
}

__global__ void zero_acc_kernel(float* __restrict__ acc, int n) {
    int i = blockIdx.x * blockDim.x + threadIdx.x;
    if (i < n) acc[i] = 0.f;
}

__global__ __launch_bounds__(BLOCK)
void edge_mlp_kernel(const float* __restrict__ coords,
                     const float* __restrict__ feat,
                     const float* __restrict__ nf,
                     const int*   __restrict__ addr_from,
                     const int*   __restrict__ addr_to,
                     const float* __restrict__ fW0, const float* __restrict__ fb0,
                     const float* __restrict__ fW1, const float* __restrict__ fb1,
                     const float* __restrict__ fW2, const float* __restrict__ fb2,
                     const float* __restrict__ tW0, const float* __restrict__ tb0,
                     const float* __restrict__ tW1, const float* __restrict__ tb1,
                     const float* __restrict__ tW2, const float* __restrict__ tb2,
                     float* __restrict__ acc)
{
    // Transposed f16 weights in LDS: Wt[n][k]  (B-fragment reads become contiguous)
    __shared__ _Float16 sW0[2][64*64];   // layer0, K padded 40->64
    __shared__ _Float16 sW1[2][64*64];   // layer1
    __shared__ _Float16 sW2[2][16*64];   // layer2 (16 outputs x 64 K)
    __shared__ float sB0[2][64], sB1[2][64], sB2[2][16];
    __shared__ _Float16 sX[WPB][16*64];  // per-wave input tile
    __shared__ _Float16 sH[WPB][16*64];  // per-wave hidden tile

    const int tid = threadIdx.x;

    // ---- cooperative weight staging (f32 global -> transposed f16 LDS) ----
    for (int i = tid; i < 64*64; i += BLOCK) {
        int n = i >> 6, k = i & 63;
        sW0[0][i] = (_Float16)((k < D_IN) ? fW0[k*64 + n] : 0.f);
        sW0[1][i] = (_Float16)((k < D_IN) ? tW0[k*64 + n] : 0.f);
        sW1[0][i] = (_Float16)fW1[k*64 + n];
        sW1[1][i] = (_Float16)tW1[k*64 + n];
    }
    for (int i = tid; i < 16*64; i += BLOCK) {
        int n = i >> 6, k = i & 63;
        sW2[0][i] = (_Float16)fW2[k*16 + n];
        sW2[1][i] = (_Float16)tW2[k*16 + n];
    }
    if (tid < 64) {
        sB0[0][tid] = fb0[tid]; sB0[1][tid] = tb0[tid];
        sB1[0][tid] = fb1[tid]; sB1[1][tid] = tb1[tid];
    }
    if (tid < 16) { sB2[0][tid] = fb2[tid]; sB2[1][tid] = tb2[tid]; }
    __syncthreads();

    const int lane  = tid & 31;
    const int wave  = tid >> 5;
    const int half  = lane >> 4;   // 0: lanes 0-15, 1: lanes 16-31
    const int l16   = lane & 15;
    const int waveId = blockIdx.x * WPB + wave;
    const int nWaves = gridDim.x * WPB;
    const int nTiles = N_EDGES / 16;

    _Float16* Xt = sX[wave];
    _Float16* Ht = sH[wave];

    // One-time: zero the constant pad region K in [40,64) of this wave's X tile.
    {
        uint4 z = {0u,0u,0u,0u};
        uint4* zp = (uint4*)(Xt + l16 * 64 + 40);   // 48B, 16B-aligned (40*2=80B)
        if (half == 1) { zp[0] = z; zp[1] = z; zp[2] = z; }
    }

    for (int tile = waveId; tile < nTiles; tile += nWaves) {
        const int e0 = tile * 16;

        // ---- stage masked input tile X[16][64] (f16), vectorized ----
        {
            const int e = e0 + l16;
            const float msk = nf[e];
            if (half == 0) {
                // K 0..31 : 8 feats + 16 "from" coords + first 8 "to" coords
                const float4* fp = (const float4*)(feat + (long)e * N_FEAT);
                const float4* cf = (const float4*)(coords + (long)addr_from[e] * D_COORD);
                const float4* ct = (const float4*)(coords + (long)addr_to[e]   * D_COORD);
                float4 f0 = fp[0], f1 = fp[1];
                float4 c0 = cf[0], c1 = cf[1], c2 = cf[2], c3 = cf[3];
                float4 t0 = ct[0], t1 = ct[1];
                uint4 w0 = { pk2(f0.x*msk, f0.y*msk), pk2(f0.z*msk, f0.w*msk),
                             pk2(f1.x*msk, f1.y*msk), pk2(f1.z*msk, f1.w*msk) };
                uint4 w1 = { pk2(c0.x*msk, c0.y*msk), pk2(c0.z*msk, c0.w*msk),
                             pk2(c1.x*msk, c1.y*msk), pk2(c1.z*msk, c1.w*msk) };
                uint4 w2 = { pk2(c2.x*msk, c2.y*msk), pk2(c2.z*msk, c2.w*msk),
                             pk2(c3.x*msk, c3.y*msk), pk2(c3.z*msk, c3.w*msk) };
                uint4 w3 = { pk2(t0.x*msk, t0.y*msk), pk2(t0.z*msk, t0.w*msk),
                             pk2(t1.x*msk, t1.y*msk), pk2(t1.z*msk, t1.w*msk) };
                uint4* dst = (uint4*)(Xt + l16 * 64);
                dst[0] = w0; dst[1] = w1; dst[2] = w2; dst[3] = w3;
            } else {
                // K 32..39 : last 8 "to" coords (K 40..63 pre-zeroed)
                const float4* ct = (const float4*)(coords + (long)addr_to[e] * D_COORD);
                float4 t2 = ct[2], t3 = ct[3];
                uint4 w0 = { pk2(t2.x*msk, t2.y*msk), pk2(t2.z*msk, t2.w*msk),
                             pk2(t3.x*msk, t3.y*msk), pk2(t3.z*msk, t3.w*msk) };
                *(uint4*)(Xt + l16 * 64 + 32) = w0;
            }
        }
        // Wave-private tile; LDS pipeline is in-order within a wave -> no barrier.

        #pragma unroll 1
        for (int m = 0; m < 2; ++m) {
            const _Float16* W0 = sW0[m];
            const _Float16* W1 = sW1[m];
            const _Float16* W2 = sW2[m];

            // ---- Layer 0: X(16x64) @ W0 -> H(16x64) ----
            v8f c0[4];
            #pragma unroll
            for (int nt = 0; nt < 4; ++nt) {
                v8f c = {0.f,0.f,0.f,0.f,0.f,0.f,0.f,0.f};
                #pragma unroll
                for (int ks = 0; ks < 2; ++ks) {
                    const int koff = ks * 32;
                    v16h a = loadA(Xt + l16 * 64, koff + half * 16);
                    v16h b = *(const v16h*)(W0 + (nt*16 + l16) * 64 + koff + half * 16);
                    c = __builtin_amdgcn_wmma_f32_16x16x32_f16(false, a, false, b,
                                                               (short)0, c, false, false);
                }
                c0[nt] = c;
            }
            #pragma unroll
            for (int nt = 0; nt < 4; ++nt) {
                const int n = nt*16 + l16;
                const float bn = sB0[m][n];
                #pragma unroll
                for (int r = 0; r < 8; ++r) {
                    float v = c0[nt][r] + bn;
                    Ht[(r + half*8) * 64 + n] = (_Float16)(v > 0.f ? v : 0.f);
                }
            }

            // ---- Layer 1: H(16x64) @ W1 -> H2(16x64) ----
            v8f c1[4];
            #pragma unroll
            for (int nt = 0; nt < 4; ++nt) {
                v8f c = {0.f,0.f,0.f,0.f,0.f,0.f,0.f,0.f};
                #pragma unroll
                for (int ks = 0; ks < 2; ++ks) {
                    const int koff = ks * 32;
                    v16h a = loadA(Ht + l16 * 64, koff + half * 16);
                    v16h b = *(const v16h*)(W1 + (nt*16 + l16) * 64 + koff + half * 16);
                    c = __builtin_amdgcn_wmma_f32_16x16x32_f16(false, a, false, b,
                                                               (short)0, c, false, false);
                }
                c1[nt] = c;
            }
            // all reads of Ht finished above -> safe to overwrite
            #pragma unroll
            for (int nt = 0; nt < 4; ++nt) {
                const int n = nt*16 + l16;
                const float bn = sB1[m][n];
                #pragma unroll
                for (int r = 0; r < 8; ++r) {
                    float v = c1[nt][r] + bn;
                    Ht[(r + half*8) * 64 + n] = (_Float16)(v > 0.f ? v : 0.f);
                }
            }

            // ---- Layer 2: H2(16x64) @ W2 -> out(16x16) ----
            v8f c2 = {0.f,0.f,0.f,0.f,0.f,0.f,0.f,0.f};
            #pragma unroll
            for (int ks = 0; ks < 2; ++ks) {
                const int koff = ks * 32;
                v16h a = loadA(Ht + l16 * 64, koff + half * 16);
                v16h b = *(const v16h*)(W2 + l16 * 64 + koff + half * 16);
                c2 = __builtin_amdgcn_wmma_f32_16x16x32_f16(false, a, false, b,
                                                            (short)0, c2, false, false);
            }

            // ---- bias + mask + scatter-add ----
            const int* __restrict__ idx = (m == 0) ? addr_from : addr_to;
            const float b2n = sB2[m][l16];
            #pragma unroll
            for (int r = 0; r < 8; ++r) {
                const int e = e0 + r + half * 8;
                const float v = (c2[r] + b2n) * nf[e];
                atomicAdd(acc + (long)idx[e] * D_OUT + l16, v);
            }
        }
    }
}

__global__ void tanh_kernel(const float* __restrict__ acc, float* __restrict__ out, int n) {
    int i = blockIdx.x * blockDim.x + threadIdx.x;
    if (i < n) out[i] = tanhf(acc[i]);
}

extern "C" void kernel_launch(void* const* d_in, const int* in_sizes, int n_in,
                              void* d_out, int out_size, void* d_ws, size_t ws_size,
                              hipStream_t stream) {
    (void)in_sizes; (void)n_in; (void)ws_size; (void)out_size;
    const float* coords    = (const float*)d_in[0];
    const float* feat      = (const float*)d_in[1];
    const float* nf        = (const float*)d_in[2];
    const int*   addr_from = (const int*)d_in[3];
    const int*   addr_to   = (const int*)d_in[4];
    const float* fW0 = (const float*)d_in[5];  const float* fb0 = (const float*)d_in[6];
    const float* fW1 = (const float*)d_in[7];  const float* fb1 = (const float*)d_in[8];
    const float* fW2 = (const float*)d_in[9];  const float* fb2 = (const float*)d_in[10];
    const float* tW0 = (const float*)d_in[11]; const float* tb0 = (const float*)d_in[12];
    const float* tW1 = (const float*)d_in[13]; const float* tb1 = (const float*)d_in[14];
    const float* tW2 = (const float*)d_in[15]; const float* tb2 = (const float*)d_in[16];

    float* acc = (float*)d_ws;                       // N_NODES * D_OUT floats
    float* out = (float*)d_out;
    const int accN = N_NODES * D_OUT;

    zero_acc_kernel<<<(accN + 255) / 256, 256, 0, stream>>>(acc, accN);

    const int nBlocks = 1024;                        // grid-stride over 50K tiles
    edge_mlp_kernel<<<nBlocks, BLOCK, 0, stream>>>(
        coords, feat, nf, addr_from, addr_to,
        fW0, fb0, fW1, fb1, fW2, fb2,
        tW0, tb0, tW1, tb1, tW2, tb2,
        acc);

    tanh_kernel<<<(accN + 255) / 256, 256, 0, stream>>>(acc, out, accN);
}
